// Top1MoE_21878563406428
// MI455X (gfx1250) — matrix operand order repaired and checked by reading, not compiled, not run on hardware
//
#include <hip/hip_runtime.h>
#include <math.h>

#define B_ 2
#define T_ 1024
#define D_ 1024
#define H_ 4096
#define E_ 8
#define N_ (B_ * T_)
#define M_ 32             // tokens per block (2 WMMA row tiles)
#define MAXT_ (N_ / M_)   // 64 token tiles worst case
#define HC_ 512           // H chunk held in LDS
#define NCH_ (H_ / HC_)   // 8 chunks

typedef __attribute__((ext_vector_type(16))) __bf16 v16bf;
typedef __attribute__((ext_vector_type(8)))  __bf16 v8bf;
typedef __attribute__((ext_vector_type(4)))  __bf16 v4bf;
typedef __attribute__((ext_vector_type(8)))  float  v8f;
typedef __attribute__((ext_vector_type(4)))  float  v4f;

__device__ __forceinline__ v16bf mk16(v8bf lo, v8bf hi) {
  v16bf r;
#pragma unroll
  for (int i = 0; i < 8; ++i) { r[i] = lo[i]; r[8 + i] = hi[i]; }
  return r;
}

__device__ __forceinline__ float gelu_exact(float v) {
  return 0.5f * v * (1.0f + erff(v * 0.70710678118654752f));
}

// async copy of 16 bytes: global -> LDS (tracked by ASYNCcnt)
__device__ __forceinline__ void async_g2l_b128(unsigned lds_off, const float* g) {
  asm volatile("global_load_async_to_lds_b128 %0, %1, off"
               :: "v"(lds_off), "v"((unsigned long long)(size_t)g)
               : "memory");
}

// ---------------- kernel 0: zero the small ws counters ----------------
__global__ __launch_bounds__(64) void zero_ws_kernel(int* __restrict__ p) {
  if (threadIdx.x < 24) p[threadIdx.x] = 0;  // counts[8], cursor[8], offsets[8]
}

// ---------------- kernel 1: gating (one wave per token) ----------------
__global__ __launch_bounds__(256) void gate_kernel(
    const float* __restrict__ x, const float* __restrict__ gw,
    const float* __restrict__ gb, int* __restrict__ expert_id,
    int* __restrict__ counts) {
  const int wav = threadIdx.x >> 5;
  const int lane = threadIdx.x & 31;
  const int n = blockIdx.x * 8 + wav;
  if (n >= N_) return;

  float acc[E_];
#pragma unroll
  for (int e = 0; e < E_; ++e) acc[e] = 0.0f;

  const float* xr = x + (size_t)n * D_;
  for (int k = lane; k < D_; k += 32) {
    const float xv = xr[k];
    const float* g = gw + (size_t)k * E_;
    v4f g0 = *(const v4f*)g;
    v4f g1 = *(const v4f*)(g + 4);
#pragma unroll
    for (int e = 0; e < 4; ++e) {
      acc[e] += xv * g0[e];
      acc[4 + e] += xv * g1[e];
    }
  }
#pragma unroll
  for (int off = 16; off >= 1; off >>= 1) {
#pragma unroll
    for (int e = 0; e < E_; ++e) acc[e] += __shfl_xor(acc[e], off, 32);
  }
  if (lane == 0) {
    float best = acc[0] + gb[0];
    int bi = 0;
#pragma unroll
    for (int e = 1; e < E_; ++e) {
      float v = acc[e] + gb[e];
      if (v > best) { best = v; bi = e; }   // first-max wins, like jnp.argmax
    }
    expert_id[n] = bi;
    atomicAdd(&counts[bi], 1);
  }
}

// ---------------- kernel 2: offsets + aux loss ----------------
__global__ void scan_kernel(const int* __restrict__ counts,
                            int* __restrict__ offsets,
                            float* __restrict__ out) {
  if (threadIdx.x == 0) {
    int off = 0;
    for (int e = 0; e < E_; ++e) { offsets[e] = off; off += counts[e]; }
    float aux = 0.0f;
    for (int e = 0; e < E_; ++e) {
      float fr = (float)counts[e] / (float)N_;  // sum(counts)==N always
      float d = fr - (1.0f / (float)E_);
      aux += d * d;
    }
    out[(size_t)N_ * D_] = aux / (float)E_;
  }
}

// ---------------- kernel 3: build per-expert token lists ----------------
__global__ __launch_bounds__(256) void scatter_kernel(
    const int* __restrict__ expert_id, const int* __restrict__ offsets,
    int* __restrict__ cursor, int* __restrict__ order) {
  const int n = blockIdx.x * 256 + threadIdx.x;
  if (n >= N_) return;
  const int e = expert_id[n];
  const int pos = offsets[e] + atomicAdd(&cursor[e], 1);
  order[pos] = n;
}

// ---------------- kernel 4: expert FFN, bf16 WMMA ----------------
__global__ __launch_bounds__(256) void moe_ffn_kernel(
    const float* __restrict__ x,
    const float* __restrict__ w1, const float* __restrict__ b1,
    const float* __restrict__ w2, const float* __restrict__ b2,
    const int* __restrict__ order, const int* __restrict__ counts,
    const int* __restrict__ offsets, float* __restrict__ out) {
  const int e = blockIdx.y;
  const int tile = blockIdx.x;
  const int cnt = counts[e];
  if (tile * M_ >= cnt) return;

  __shared__ int    toks[M_];
  __shared__ float  Xf[8][D_];        // 32 KB: async fp32 staging (8 rows/pass)
  __shared__ __bf16 Xs[M_][D_];       // 64 KB: token tile, bf16
  __shared__ __bf16 Hs[M_][HC_];      // 32 KB: gelu(x@w1) chunk, bf16
  __shared__ __bf16 Wst[8][16][32];   // 8 KB: per-wave transposed weight tile

  const int tid = threadIdx.x;
  const int wave = tid >> 5;
  const int lane = tid & 31;
  const int rc = lane & 15;   // WMMA row (A) / column (B,C,D) of this lane
  const int half = lane >> 4;
  const int rg = lane & 7;    // staging: 4 k-rows owned
  const int cg = lane >> 3;   // staging: 4 cols owned

  if (tid < M_) {
    const int idx = tile * M_ + tid;
    toks[tid] = (idx < cnt) ? order[offsets[e] + idx] : -1;
  }
  __syncthreads();

  // ---- stage X tile -> LDS via async copies, then convert to bf16 ----
  for (int p = 0; p < M_ / 8; ++p) {
    const int r = tid >> 5;           // one row per wave
    const int cs = (tid & 31) * 32;   // 32 floats per lane
    const int grow = p * 8 + r;
    const int tk = toks[grow];
    if (tk >= 0) {
      const float* src = x + (size_t)tk * D_ + cs;
      const unsigned l0 = (unsigned)(size_t)&Xf[r][cs];
#pragma unroll
      for (int i = 0; i < 8; ++i) async_g2l_b128(l0 + i * 16u, src + i * 4);
    }
    asm volatile("s_wait_asynccnt 0" ::: "memory");
    __syncthreads();
    if (tk >= 0) {
#pragma unroll
      for (int i = 0; i < 8; ++i) {
        v4f v = *(const v4f*)&Xf[r][cs + i * 4];
        v4bf pb;
#pragma unroll
        for (int j = 0; j < 4; ++j) pb[j] = (__bf16)v[j];
        *(v4bf*)&Xs[grow][cs + i * 4] = pb;
      }
    } else {
      v4bf z;
#pragma unroll
      for (int j = 0; j < 4; ++j) z[j] = (__bf16)0.0f;
#pragma unroll
      for (int i = 0; i < 8; ++i) *(v4bf*)&Xs[grow][cs + i * 4] = z;
    }
    __syncthreads();
  }

  const v8f vzero = {0.f, 0.f, 0.f, 0.f, 0.f, 0.f, 0.f, 0.f};
  v8f y0[8], y1[8];
#pragma unroll
  for (int t = 0; t < 8; ++t) { y0[t] = vzero; y1[t] = vzero; }

  const float* W1e = w1 + (size_t)e * D_ * H_;
  const float* W2e = w2 + (size_t)e * H_ * D_;

  for (int c = 0; c < NCH_; ++c) {
    // ---- phase 1: Hs = gelu(X @ W1[:, chunk] + b1) ----
#pragma unroll 1
    for (int tt = 0; tt < HC_ / 16 / 8; ++tt) {
      const int lt = wave + tt * 8;           // local 16-col tile in chunk
      const int colbase = c * HC_ + lt * 16;  // column in H
      v8f acc0 = vzero, acc1 = vzero;
#pragma unroll 1
      for (int k0 = 0; k0 < D_; k0 += 32) {
        // coalesced 4x16B load of W1 rows, bf16-pack, b64 transposed stores
        const float* wr = W1e + (size_t)(k0 + 4 * rg) * H_ + colbase + 4 * cg;
        v4f r0 = *(const v4f*)(wr);
        v4f r1 = *(const v4f*)(wr + H_);
        v4f r2 = *(const v4f*)(wr + 2 * H_);
        v4f r3 = *(const v4f*)(wr + 3 * H_);
        const float* pf = (k0 + 32 < D_) ? (wr + 32 * (size_t)H_) : wr;
        __builtin_prefetch((const void*)pf, 0, 1);
#pragma unroll
        for (int cI = 0; cI < 4; ++cI) {
          v4bf pq;
          pq[0] = (__bf16)r0[cI]; pq[1] = (__bf16)r1[cI];
          pq[2] = (__bf16)r2[cI]; pq[3] = (__bf16)r3[cI];
          *(v4bf*)&Wst[wave][4 * cg + cI][4 * rg] = pq;
        }
        __builtin_amdgcn_wave_barrier();
        v8bf a0lo = *(const v8bf*)&Xs[rc][k0 + half * 8];
        v8bf a0hi = *(const v8bf*)&Xs[rc][k0 + 16 + half * 8];
        v8bf a1lo = *(const v8bf*)&Xs[16 + rc][k0 + half * 8];
        v8bf a1hi = *(const v8bf*)&Xs[16 + rc][k0 + 16 + half * 8];
        const __bf16* bp = &Wst[wave][rc][half * 8];
        v16bf bfr = mk16(*(const v8bf*)bp, *(const v8bf*)(bp + 16));
        acc0 = __builtin_amdgcn_wmma_f32_16x16x32_bf16(
            false, mk16(a0lo, a0hi), false, bfr, (short)0, acc0, false, false);
        acc1 = __builtin_amdgcn_wmma_f32_16x16x32_bf16(
            false, mk16(a1lo, a1hi), false, bfr, (short)0, acc1, false, false);
        __builtin_amdgcn_wave_barrier();
      }
      const float bb = b1[(size_t)e * H_ + colbase + rc];
#pragma unroll
      for (int r = 0; r < 8; ++r) {
        Hs[half * 8 + r][lt * 16 + rc]      = (__bf16)gelu_exact(acc0[r] + bb);
        Hs[16 + half * 8 + r][lt * 16 + rc] = (__bf16)gelu_exact(acc1[r] + bb);
      }
    }
    __syncthreads();

    // ---- phase 2: Y += Hs @ W2[chunk rows, :] ----
#pragma unroll
    for (int t2 = 0; t2 < 8; ++t2) {
      const int colbase = wave * 128 + t2 * 16;  // column in D, owned by wave
      v8f acc0 = y0[t2], acc1 = y1[t2];
#pragma unroll 1
      for (int k0 = 0; k0 < HC_; k0 += 32) {
        const float* wr =
            W2e + (size_t)(c * HC_ + k0 + 4 * rg) * D_ + colbase + 4 * cg;
        v4f r0 = *(const v4f*)(wr);
        v4f r1 = *(const v4f*)(wr + D_);
        v4f r2 = *(const v4f*)(wr + 2 * D_);
        v4f r3 = *(const v4f*)(wr + 3 * D_);
        const float* pf = (k0 + 32 < HC_) ? (wr + 32 * (size_t)D_) : wr;
        __builtin_prefetch((const void*)pf, 0, 1);
#pragma unroll
        for (int cI = 0; cI < 4; ++cI) {
          v4bf pq;
          pq[0] = (__bf16)r0[cI]; pq[1] = (__bf16)r1[cI];
          pq[2] = (__bf16)r2[cI]; pq[3] = (__bf16)r3[cI];
          *(v4bf*)&Wst[wave][4 * cg + cI][4 * rg] = pq;
        }
        __builtin_amdgcn_wave_barrier();
        v8bf a0lo = *(const v8bf*)&Hs[rc][k0 + half * 8];
        v8bf a0hi = *(const v8bf*)&Hs[rc][k0 + 16 + half * 8];
        v8bf a1lo = *(const v8bf*)&Hs[16 + rc][k0 + half * 8];
        v8bf a1hi = *(const v8bf*)&Hs[16 + rc][k0 + 16 + half * 8];
        const __bf16* bp = &Wst[wave][rc][half * 8];
        v16bf bfr = mk16(*(const v8bf*)bp, *(const v8bf*)(bp + 16));
        acc0 = __builtin_amdgcn_wmma_f32_16x16x32_bf16(
            false, mk16(a0lo, a0hi), false, bfr, (short)0, acc0, false, false);
        acc1 = __builtin_amdgcn_wmma_f32_16x16x32_bf16(
            false, mk16(a1lo, a1hi), false, bfr, (short)0, acc1, false, false);
        __builtin_amdgcn_wave_barrier();
      }
      y0[t2] = acc0;
      y1[t2] = acc1;
    }
    __syncthreads();
  }

  // ---- epilogue: + b2, scatter rows back to token positions ----
#pragma unroll
  for (int t2 = 0; t2 < 8; ++t2) {
    const int col = wave * 128 + t2 * 16 + rc;
    const float bb2 = b2[(size_t)e * D_ + col];
#pragma unroll
    for (int r = 0; r < 8; ++r) {
      const int row = half * 8 + r;
      const int tk0 = toks[row];
      const int tk1 = toks[16 + row];
      if (tk0 >= 0) out[(size_t)tk0 * D_ + col] = y0[t2][r] + bb2;
      if (tk1 >= 0) out[(size_t)tk1 * D_ + col] = y1[t2][r] + bb2;
    }
  }
}

// ---------------- launch ----------------
extern "C" void kernel_launch(void* const* d_in, const int* in_sizes, int n_in,
                              void* d_out, int out_size, void* d_ws,
                              size_t ws_size, hipStream_t stream) {
  (void)in_sizes; (void)n_in; (void)out_size; (void)ws_size;
  const float* x      = (const float*)d_in[0];
  const float* gate_w = (const float*)d_in[1];
  const float* gate_b = (const float*)d_in[2];
  const float* w1     = (const float*)d_in[3];
  const float* b1     = (const float*)d_in[4];
  const float* w2     = (const float*)d_in[5];
  const float* b2     = (const float*)d_in[6];
  float* out = (float*)d_out;

  int* counts    = (int*)d_ws;        // [8]
  int* cursor    = counts + 8;        // [8]
  int* offsets   = counts + 16;       // [8]
  int* expert_id = counts + 64;       // [N]
  int* order     = expert_id + N_;    // [N]

  zero_ws_kernel<<<1, 64, 0, stream>>>(counts);
  gate_kernel<<<N_ / 8, 256, 0, stream>>>(x, gate_w, gate_b, expert_id, counts);
  scan_kernel<<<1, 32, 0, stream>>>(counts, offsets, out);
  scatter_kernel<<<N_ / 256, 256, 0, stream>>>(expert_id, offsets, cursor, order);
  moe_ffn_kernel<<<dim3(MAXT_, E_), 256, 0, stream>>>(
      x, w1, b1, w2, b2, order, counts, offsets, out);
}
